// CPRINT4Linear_64330020159893
// MI455X (gfx1250) — matrix operand order repaired and checked by reading, not compiled, not run on hardware
//
#include <hip/hip_runtime.h>

typedef _Float16 v16h __attribute__((ext_vector_type(16)));
typedef _Float16 v8h  __attribute__((ext_vector_type(8)));
typedef _Float16 v2h  __attribute__((ext_vector_type(2)));
typedef float    v8f  __attribute__((ext_vector_type(8)));

#define K_DIM   4096
#define N_DIM   11008
#define GROUP   128
#define BM      128
#define BN      128
#define BK      32
#define LDA     40    // padded LDS row stride in halves (80B) - bank spread, 16B aligned
#define LDB     40

__global__ __launch_bounds__(256, 1)
void w4a16_gather_wmma_kernel(const float* __restrict__ x,
                              const int*   __restrict__ col_indices,
                              const int*   __restrict__ w_packed,
                              const float* __restrict__ scales,
                              const float* __restrict__ bias,
                              float*       __restrict__ out)
{
    __shared__ _Float16 As[BM * LDA];   // [m][k], k contiguous
    __shared__ _Float16 Bs[BN * LDB];   // [n][k], k contiguous (B stored N-major)

    const int tid   = threadIdx.x;
    const int lane  = tid & 31;
    const int wave  = tid >> 5;
    const int waveM = wave & 3;    // 4 waves along M, 32 rows each
    const int waveN = wave >> 2;   // 2 waves along N, 64 cols each

    const int nBase = blockIdx.x * BN;
    const int mBase = blockIdx.y * BM;

    v8f acc[2][4];
#pragma unroll
    for (int mi = 0; mi < 2; ++mi)
#pragma unroll
        for (int ni = 0; ni < 4; ++ni) {
            v8f z = {0.f, 0.f, 0.f, 0.f, 0.f, 0.f, 0.f, 0.f};
            acc[mi][ni] = z;
        }

    // A staging geometry: thread covers rows aM0 + 32j (j=0..3), 4 consecutive k
    const int aM0 = tid >> 3;          // 0..31
    const int aK0 = (tid & 7) * 4;     // 0,4,...,28
    // B staging geometry: thread owns one n column, k-pair rows bKp0 + 2i (i=0..7)
    const int bN   = tid & 127;        // 0..127
    const int bKp0 = tid >> 7;         // 0..1

    const int fragRow = lane & 15;         // M or N within 16x16 tile
    const int aHalfK  = (lane >> 4) * 8;   // A: K run base (0 or 8), second run +16
    const int bHalfK  = (lane >> 4) * 8;   // B: K run base (0 or 8), second run +16

    for (int kt = 0; kt < K_DIM / BK; ++kt) {
        const int kBase = kt * BK;

        // ---- stage A: fused column-permutation gather, fp32 -> f16 ----
        int idx0 = col_indices[kBase + aK0 + 0];
        int idx1 = col_indices[kBase + aK0 + 1];
        int idx2 = col_indices[kBase + aK0 + 2];
        int idx3 = col_indices[kBase + aK0 + 3];
#pragma unroll
        for (int j = 0; j < 4; ++j) {
            const int mloc = aM0 + j * 32;
            const float* xrow = x + (size_t)(mBase + mloc) * K_DIM;
            _Float16* arow = &As[mloc * LDA + aK0];
            arow[0] = (_Float16)xrow[idx0];
            arow[1] = (_Float16)xrow[idx1];
            arow[2] = (_Float16)xrow[idx2];
            arow[3] = (_Float16)xrow[idx3];
        }

        // ---- stage B: int4 unpack + group dequant, write N-major f16 ----
        const int   grp    = kBase / GROUP;    // constant over this K-step
        const float s      = scales[(size_t)grp * N_DIM + nBase + bN];
        const int   kpBase = kBase >> 1;
#pragma unroll
        for (int i = 0; i < 8; ++i) {
            const int kp = bKp0 + 2 * i;
            const int w  = w_packed[(size_t)(kpBase + kp) * N_DIM + nBase + bN];
            v2h pk;
            pk.x = (_Float16)((float)((w & 15) - 8) * s);         // even k
            pk.y = (_Float16)((float)(((w >> 4) & 15) - 8) * s);  // odd k
            *(v2h*)&Bs[bN * LDB + 2 * kp] = pk;
        }

        __syncthreads();

        // ---- A fragments (ISA 16-bit A layout: two 8-half runs per lane) ----
        v16h afrag[2];
#pragma unroll
        for (int mi = 0; mi < 2; ++mi) {
            const _Float16* ap = &As[(waveM * 32 + mi * 16 + fragRow) * LDA + aHalfK];
            v8h alo = *(const v8h*)ap;
            v8h ahi = *(const v8h*)(ap + 16);
            afrag[mi] = __builtin_shufflevector(alo, ahi,
                0, 1, 2, 3, 4, 5, 6, 7, 8, 9, 10, 11, 12, 13, 14, 15);
        }

        // ---- B fragments + WMMA ----
#pragma unroll
        for (int ni = 0; ni < 4; ++ni) {
            const _Float16* bp = &Bs[(waveN * 64 + ni * 16 + fragRow) * LDB + bHalfK];
            v8h blo = *(const v8h*)bp;
            v8h bhi = *(const v8h*)(bp + 16);
            v16h bfrag = __builtin_shufflevector(blo, bhi,
                0, 1, 2, 3, 4, 5, 6, 7, 8, 9, 10, 11, 12, 13, 14, 15);
#pragma unroll
            for (int mi = 0; mi < 2; ++mi) {
                acc[mi][ni] = __builtin_amdgcn_wmma_f32_16x16x32_f16(
                    /*neg_a=*/false, afrag[mi],
                    /*neg_b=*/false, bfrag,
                    /*c_mod=*/(short)0, acc[mi][ni],
                    /*reuse_a=*/false, /*reuse_b=*/false);
            }
        }

        __syncthreads();
    }

    // ---- epilogue: f32 C/D layout -> global, add bias ----
    const int rowHi = (lane >> 4) * 8;   // VGPR v holds M = rowHi + v
#pragma unroll
    for (int ni = 0; ni < 4; ++ni) {
        const int col = nBase + waveN * 64 + ni * 16 + fragRow;
        const float bv = bias[col];
#pragma unroll
        for (int mi = 0; mi < 2; ++mi) {
            const int row0 = mBase + waveM * 32 + mi * 16 + rowHi;
#pragma unroll
            for (int v = 0; v < 8; ++v) {
                out[(size_t)(row0 + v) * N_DIM + col] = acc[mi][ni][v] + bv;
            }
        }
    }
}

extern "C" void kernel_launch(void* const* d_in, const int* in_sizes, int n_in,
                              void* d_out, int out_size, void* d_ws, size_t ws_size,
                              hipStream_t stream) {
    const float* x           = (const float*)d_in[0];
    const int*   col_indices = (const int*)  d_in[1];
    const int*   w_packed    = (const int*)  d_in[2];
    const float* scales      = (const float*)d_in[3];
    const float* bias        = (const float*)d_in[4];
    float*       out         = (float*)d_out;

    const int M = in_sizes[0] / K_DIM;   // B*S = 8192
    dim3 grid(N_DIM / BN, M / BM);       // (86, 64)
    dim3 block(256);
    w4a16_gather_wmma_kernel<<<grid, block, 0, stream>>>(
        x, col_indices, w_packed, scales, bias, out);
}
